// EmbedMetadata_51135880626864
// MI455X (gfx1250) — matrix operand order repaired and checked by reading, not compiled, not run on hardware
//
#include <hip/hip_runtime.h>
#include <hip/hip_bf16.h>

typedef __attribute__((ext_vector_type(16))) _Float16 v16h;
typedef __attribute__((ext_vector_type(8)))  _Float16 v8h;
typedef __attribute__((ext_vector_type(8)))  float    v8f;

#define BB   16384
#define II   35
#define EE   256
#define CC   64
#define KDIM 256   // 245 real + 11 zero pad
#define ROWS 32    // batch rows per block

// ---------------------------------------------------------------------------
// Kernel 1: fold Wd/bd/Wc/bc/Wr/br/emb/Wf/bf into a 256x256 f16 matrix M
// (stored in WMMA B-fragment order) and a 256-float const vector.
//   M[k,e]:  k<35: A_d ; 35..69: A_c ; 70..104: A_r ; 105..244: G[i,t,e]
//   const[e] = bf[e] + sum_i (bd.Wf + bc.Wf + br.Wf)
// B-fragment layout used here for v_wmma_f32_16x16x32_f16, contiguous K, same order as A (32x16 K-slice):
//   lane<16 : col = lane,    elems j=0..15 -> K = j
//   lane>=16: col = lane-16, elems j=0..15 -> K = 16+j
// fragment id = (nTile*8 + kstep), each fragment = 32 lanes * 16 halfs.
// ---------------------------------------------------------------------------
__global__ __launch_bounds__(256) void fold_weights_kernel(
    const float* __restrict__ Wd, const float* __restrict__ bd,
    const float* __restrict__ Wc, const float* __restrict__ bc,
    const float* __restrict__ Wr, const float* __restrict__ br,
    const float* __restrict__ emb, const float* __restrict__ Wf,
    const float* __restrict__ bf,
    _Float16* __restrict__ Mfrag, float* __restrict__ constOut)
{
    const int e = blockIdx.x;    // 0..255 output feature
    const int k = threadIdx.x;   // 0..255 folded-K index
    const float* wfRow = Wf + (size_t)e * (II * 4 * CC);

    float dot  = 0.f;
    float bdot = 0.f;
    if (k < 245) {
        const float* src;
        const float* bsrc = nullptr;
        int wfOff;
        if (k < 35)       { int i = k;      src = Wd  + i * CC; wfOff = i * 256;       bsrc = bd + i * CC; }
        else if (k < 70)  { int i = k - 35; src = Wc  + i * CC; wfOff = i * 256 + 64;  bsrc = bc + i * CC; }
        else if (k < 105) { int i = k - 70; src = Wr  + i * CC; wfOff = i * 256 + 128; bsrc = br + i * CC; }
        else { int j = k - 105; int i = j >> 2; int t = j & 3;
               src = emb + ((size_t)i * 4 + t) * CC; wfOff = i * 256 + 192; }
        const float* wf = wfRow + wfOff;
        #pragma unroll 8
        for (int c = 0; c < CC; ++c) {
            dot += src[c] * wf[c];
            if (bsrc) bdot += bsrc[c] * wf[c];
        }
    }

    // scatter into B-fragment layout
    {
        const int n    = e >> 4;
        const int cit  = e & 15;
        const int kst  = k >> 5;
        const int kk   = k & 31;
        const int lane = (kk < 16) ? cit : (16 + cit);
        const int j    = kk & 15;
        Mfrag[(((size_t)(n * 8 + kst) * 32 + lane) << 4) + j] =
            (k < 245) ? (_Float16)dot : (_Float16)0.f;
    }

    // reduce bias contributions (k<105) into const[e]
    __shared__ float cred[256];
    cred[k] = (k < 105) ? bdot : 0.f;
    __syncthreads();
    for (int s = 128; s > 0; s >>= 1) {
        if (k < s) cred[k] += cred[k + s];
        __syncthreads();
    }
    if (k == 0) constOut[e] = cred[0] + bf[e];
}

// ---------------------------------------------------------------------------
// Kernel 2: build X tile in LDS, WMMA GEMM (32x256 @ 256x256), fused
// +const, LayerNorm, ReLU epilogue.
// Block: 256 threads = 8 wave32s. wave w: rowTile = w>>2, colGroup = w&3
// (4 N-tiles of 16 cols each -> 64 cols per wave).
// ---------------------------------------------------------------------------
__global__ __launch_bounds__(256) void gemm_ln_relu_kernel(
    const float* __restrict__ metadata,
    const float* __restrict__ gamma, const float* __restrict__ beta,
    const _Float16* __restrict__ Mfrag, const float* __restrict__ constG,
    float* __restrict__ out)
{
    __shared__ _Float16 Xs[ROWS][KDIM];    // 16 KB
    __shared__ float    outS[ROWS][EE];    // 32 KB
    __shared__ float    constS[EE];
    __shared__ float    gammaS[EE];
    __shared__ float    betaS[EE];
    __shared__ float    redS[ROWS][8];
    __shared__ float    red2S[ROWS][8];
    __shared__ float    muS[ROWS];
    __shared__ float    rsS[ROWS];

    const int tid = threadIdx.x;
    const int b0  = blockIdx.x * ROWS;

    constS[tid] = constG[tid];
    gammaS[tid] = gamma[tid];
    betaS[tid]  = beta[tid];

    // ---- build X[row][k] in f16: [depth|cov|rlen|onehot4(rt)|pad] ----
    {
        const int row = tid >> 3;          // 0..31
        const int kc  = tid & 7;           // 0..7 -> 32 k each
        const float* mrow = metadata + (size_t)(b0 + row) * (4 * II);
        #pragma unroll 4
        for (int m = 0; m < 32; ++m) {
            const int k = kc * 32 + m;
            float v;
            if (k < 105) {                 // depth/cov/rlen are contiguous
                v = mrow[k];
            } else if (k < 245) {
                const int j = k - 105;
                const int i = j >> 2;
                const int t = j & 3;
                const int r = (int)mrow[105 + i];          // in {-2,-1,0,1}
                const int rt = (r == -1) ? 2 : ((r == -2) ? 3 : r);
                v = (t == rt) ? 1.f : 0.f;
            } else {
                v = 0.f;
            }
            Xs[row][k] = (_Float16)v;
        }
    }
    __syncthreads();

    // ---- WMMA GEMM ----
    const int lane      = tid & 31;
    const int wv        = tid >> 5;
    const int rowTile   = wv >> 2;          // 0..1
    const int colGroup  = wv & 3;           // 0..3
    const int aRow      = rowTile * 16 + (lane & 15);
    const int kbaseA    = (lane < 16) ? 0 : 16;  // A-fragment K base
    const int colInTile = lane & 15;
    const int drBase    = rowTile * 16 + ((lane >> 4) << 3); // D row base

    for (int t = 0; t < 4; ++t) {
        const int n = colGroup * 4 + t;
        v8f acc = {};
        const _Float16* bbase = Mfrag + (size_t)(n * 8) * 512 + lane * 16;
        #pragma unroll
        for (int ks = 0; ks < 8; ++ks) {
            // A fragment from LDS: lane holds row aRow, K = ks*32 + kbaseA + {0..15}
            v8h lo = *(const v8h*)&Xs[aRow][ks * 32 + kbaseA];
            v8h hi = *(const v8h*)&Xs[aRow][ks * 32 + kbaseA + 8];
            v16h a;
            #pragma unroll
            for (int j = 0; j < 8; ++j) { a[j] = lo[j]; a[j + 8] = hi[j]; }
            // B fragment: contiguous 32B per lane, straight from L2-resident Mfrag
            v16h bm = *(const v16h*)(bbase + (size_t)ks * 512);
            acc = __builtin_amdgcn_wmma_f32_16x16x32_f16(
                false, a, false, bm, (short)0, acc, false, false);
        }
        const int col  = n * 16 + colInTile;
        const float ca = constS[col];
        #pragma unroll
        for (int v = 0; v < 8; ++v)
            outS[drBase + v][col] = acc[v] + ca;   // D row = drBase + vgpr
    }
    __syncthreads();

    // ---- LayerNorm + ReLU epilogue ----
    {
        const int r = tid >> 3, seg = tid & 7;
        float s = 0.f, s2 = 0.f;
        #pragma unroll 8
        for (int c = seg * 32; c < seg * 32 + 32; ++c) {
            const float v = outS[r][c];
            s += v; s2 += v * v;
        }
        redS[r][seg] = s;
        red2S[r][seg] = s2;
    }
    __syncthreads();
    if (tid < ROWS) {
        float s = 0.f, s2 = 0.f;
        #pragma unroll
        for (int q = 0; q < 8; ++q) { s += redS[tid][q]; s2 += red2S[tid][q]; }
        const float mu  = s * (1.f / EE);
        const float var = s2 * (1.f / EE) - mu * mu;
        muS[tid] = mu;
        rsS[tid] = rsqrtf(var + 1e-5f);
    }
    __syncthreads();
    {
        const int r = tid >> 3, seg = tid & 7;
        const float mu = muS[r], rs = rsS[r];
        float* op = out + (size_t)(b0 + r) * EE;
        #pragma unroll 8
        for (int c = seg * 32; c < seg * 32 + 32; ++c) {
            const float v = (outS[r][c] - mu) * rs * gammaS[c] + betaS[c];
            op[c] = fmaxf(v, 0.f);
        }
    }
}

// ---------------------------------------------------------------------------
extern "C" void kernel_launch(void* const* d_in, const int* in_sizes, int n_in,
                              void* d_out, int out_size, void* d_ws, size_t ws_size,
                              hipStream_t stream) {
    const float* metadata = (const float*)d_in[0];
    const float* Wd   = (const float*)d_in[1];
    const float* bd   = (const float*)d_in[2];
    const float* Wc   = (const float*)d_in[3];
    const float* bc   = (const float*)d_in[4];
    const float* Wr   = (const float*)d_in[5];
    const float* br   = (const float*)d_in[6];
    const float* emb  = (const float*)d_in[7];
    const float* Wf   = (const float*)d_in[8];
    const float* bf   = (const float*)d_in[9];
    const float* gamma = (const float*)d_in[10];
    const float* beta  = (const float*)d_in[11];

    _Float16* Mfrag  = (_Float16*)d_ws;                              // 128 KB
    float*    constB = (float*)((char*)d_ws + (size_t)KDIM * EE * 2); // 1 KB

    fold_weights_kernel<<<EE, 256, 0, stream>>>(
        Wd, bd, Wc, bc, Wr, br, emb, Wf, bf, Mfrag, constB);

    gemm_ln_relu_kernel<<<BB / ROWS, 256, 0, stream>>>(
        metadata, gamma, beta, Mfrag, constB, (float*)d_out);
}
